// FourierBlock_11458972745767
// MI455X (gfx1250) — compile-verified, matrix operand order checked by reading, not run on hardware
//
#include <hip/hip_runtime.h>
#include <cstdint>
#include <cstddef>

#define B_ 32
#define D_ 64
#define N_ 128
#define L_ 192
#define M_ 32

typedef __attribute__((ext_vector_type(16))) _Float16 v16h;
typedef __attribute__((ext_vector_type(8)))  _Float16 v8h;
typedef __attribute__((ext_vector_type(8)))  float    v8f;
typedef __attribute__((ext_vector_type(4)))  int      v4i;

// Workspace layout (in _Float16 elements):
//   [0, B*N*64*64)    : XY buffer [B][N][64][64]  (X written by DFT, overwritten in place by Y)
//   [FT_OFF, +64*192) : Ft [64][192]  forward DFT matrix, transposed (row j = column of F)
//   [GT_OFF, +192*64) : Gt [192][64]  inverse DFT matrix, transposed (row l = column of G)
static const size_t XY_OFF = 0;
static const size_t FT_OFF = (size_t)B_ * N_ * 64 * 64;
static const size_t GT_OFF = FT_OFF + (size_t)64 * 192;

// ---- CDNA5 async global->LDS copy path (guarded; falls back to sync copies) ----
#if defined(__has_builtin)
#if __has_builtin(__builtin_amdgcn_global_load_async_to_lds_b128) && \
    __has_builtin(__builtin_amdgcn_s_wait_asynccnt)
#define FB_ASYNC 1
#endif
#endif
#ifndef FB_ASYNC
#define FB_ASYNC 0
#endif

// Copy one 16-byte chunk global -> LDS.
__device__ __forceinline__ void copy16(const _Float16* g, _Float16* l) {
#if FB_ASYNC
  __builtin_amdgcn_global_load_async_to_lds_b128((v4i*)g, (v4i*)l, 0, 0);
#else
  *(v8h*)l = *(const v8h*)g;
#endif
}

// Wait for all outstanding async copies, then block barrier.
__device__ __forceinline__ void stage_join() {
#if FB_ASYNC
  __builtin_amdgcn_s_wait_asynccnt(0);
#endif
  __syncthreads();
}

// Load one WMMA A/B fragment: 16 halfs per lane as two contiguous 8-half (16B) chunks
// at k-offsets (base) and (base+16). Caller passes p = row_base + kbase + 8*half.
__device__ __forceinline__ v16h load_frag(const _Float16* p) {
  v8h lo = *(const v8h*)(p);
  v8h hi = *(const v8h*)(p + 16);
  return __builtin_shufflevector(lo, hi, 0,1,2,3,4,5,6,7,8,9,10,11,12,13,14,15);
}

__device__ __forceinline__ v8f wmma_f16(v16h a, v16h b, v8f c) {
  return __builtin_amdgcn_wmma_f32_16x16x32_f16(false, a, false, b, (short)0, c, false, false);
}

// ---------------------------------------------------------------------------
// Kernel 0: build twiddle tables.
//   Ft[j][l], j<32:  cos(2*pi*j*l/L)        (real part basis of rfft, e^{-i t})
//            j>=32: -sin(2*pi*(j-32)*l/L)   (imag part basis)
//   Gt[l][k], k==0:  1/L
//            1<=k<32:  (2/L)*cos(2*pi*k*l/L)
//            k==32:    0                      (irfft drops imag of mode 0)
//            k>32:    -(2/L)*sin(2*pi*(k-32)*l/L)
// ---------------------------------------------------------------------------
__global__ void fb_init_tables(_Float16* __restrict__ ws) {
  const float TWO_PI = 6.28318530717958647692f;
  const int tid = threadIdx.x;
  _Float16* Ft = ws + FT_OFF;
  _Float16* Gt = ws + GT_OFF;
  for (int idx = tid; idx < 64 * 192; idx += blockDim.x) {
    int j = idx / 192, l = idx - j * 192;
    int mm = j & 31;
    float th = TWO_PI * (float)((mm * l) % 192) * (1.0f / 192.0f);
    float v = (j < 32) ? __builtin_cosf(th) : -__builtin_sinf(th);
    Ft[idx] = (_Float16)v;
  }
  for (int idx = tid; idx < 192 * 64; idx += blockDim.x) {
    int l = idx >> 6, k = idx & 63;
    int mm = k & 31;
    float th = TWO_PI * (float)((mm * l) % 192) * (1.0f / 192.0f);
    float scale = (mm == 0) ? (1.0f / 192.0f) : (2.0f / 192.0f);
    float v;
    if (k < 32)       v = scale * __builtin_cosf(th);
    else if (k == 32) v = 0.0f;
    else              v = -scale * __builtin_sinf(th);
    Gt[idx] = (_Float16)v;
  }
}

// ---------------------------------------------------------------------------
// Kernel 1: truncated DFT.  Per block = one (b,n):
//   X[d][j] = sum_l q[b,d,n,l] * F[l][j],  [64x192]x[192x64] via 16x16x32 f16 WMMA.
// q is converted f32->f16 through VGPRs; the Ft table is async-staged to LDS.
// Result stored f16 in ws as [b][n][d][j] (j<32 real, j>=32 imag).
// ---------------------------------------------------------------------------
__global__ __launch_bounds__(128) void fb_dft(const float* __restrict__ q,
                                              _Float16* __restrict__ ws) {
  __shared__ __align__(16) _Float16 qs[64][200];   // padded: 400B row = 16B-mult
  __shared__ __align__(16) _Float16 fts[64][200];
  const int bid = blockIdx.x;
  const int n = bid & (N_ - 1);
  const int b = bid >> 7;
  const int tid = threadIdx.x;

  // Async-stage Ft (64 rows x 192 halfs = 24 chunks/row) into LDS.
  const _Float16* Ft = ws + FT_OFF;
  for (int c = tid; c < 64 * 24; c += 128) {
    int row = c / 24, col = c - row * 24;
    copy16(Ft + (size_t)row * 192 + col * 8, &fts[row][col * 8]);
  }

  // Convert-stage q slice [64][192] f32 -> f16.
  const float* qp = q + ((size_t)b * D_ * N_ + n) * L_;
  for (int idx = tid; idx < 64 * 192; idx += 128) {
    int i = idx / 192, l = idx - i * 192;
    qs[i][l] = (_Float16)qp[(size_t)i * (N_ * L_) + l];
  }
  stage_join();

  const int lane = tid & 31;
  const int wave = tid >> 5;     // d-tile, 4 waves cover d=0..63
  const int half = lane >> 4;
  const int r = lane & 15;

  v8f acc[4] = {};
  for (int kk = 0; kk < 6; ++kk) {
    const int k0 = kk * 32 + 8 * half;
    v16h a = load_frag(&qs[wave * 16 + r][k0]);
#pragma unroll
    for (int jt = 0; jt < 4; ++jt) {
      v16h bb = load_frag(&fts[jt * 16 + r][k0]);
      acc[jt] = wmma_f16(a, bb, acc[jt]);
    }
  }

  _Float16* Xp = ws + XY_OFF + (size_t)(b * N_ + n) * 4096;
#pragma unroll
  for (int jt = 0; jt < 4; ++jt)
#pragma unroll
    for (int v = 0; v < 8; ++v) {
      int d = wave * 16 + 8 * half + v;
      int j = jt * 16 + r;
      Xp[d * 64 + j] = (_Float16)acc[jt][v];
    }
}

// ---------------------------------------------------------------------------
// Kernel 2: complex mode mix.  Per block = one (n,m):
//   Yr[b][o] = Xr*Wr - Xi*Wi ; Yi[b][o] = Xr*Wi + Xi*Wr   ([32x64]x[64x64])
// Writes in place over X (this block exclusively owns columns m, m+32 of node n).
// ---------------------------------------------------------------------------
__global__ __launch_bounds__(128) void fb_modemix(const float* __restrict__ w_real,
                                                  const float* __restrict__ w_imag,
                                                  _Float16* __restrict__ ws) {
  __shared__ __align__(16) _Float16 xr[32][72], xi[32][72];
  __shared__ __align__(16) _Float16 wr[64][72], wi[64][72], wni[64][72];
  const int m = blockIdx.x;
  const int n = blockIdx.y;
  const int tid = threadIdx.x;
  _Float16* XY = ws + XY_OFF;

  for (int idx = tid; idx < 2048; idx += 128) {
    int b = idx >> 6, i = idx & 63;
    size_t base = ((size_t)(b * N_ + n) * 64 + i) * 64;
    xr[b][i] = XY[base + m];
    xi[b][i] = XY[base + m + 32];
  }
  const float* wrp = w_real + (size_t)n * 64 * 64 * 32;
  const float* wip = w_imag + (size_t)n * 64 * 64 * 32;
  for (int idx = tid; idx < 4096; idx += 128) {
    int i = idx >> 6, o = idx & 63;
    size_t off = ((size_t)i * 64 + o) * 32 + m;
    float wrv = wrp[off];
    float wiv = wip[off];
    wr[o][i]  = (_Float16)wrv;      // transposed: B[k=i][n=o] stored as [o][i]
    wi[o][i]  = (_Float16)wiv;
    wni[o][i] = (_Float16)(-wiv);   // negated copy avoids relying on neg_a modifier
  }
  __syncthreads();

  const int lane = tid & 31;
  const int wave = tid >> 5;
  const int btile = wave >> 1;     // b rows 0-15 / 16-31
  const int og = wave & 1;         // o cols 0-31 / 32-63
  const int half = lane >> 4;
  const int r = lane & 15;

  v8f accr[2] = {}, acci[2] = {};
  for (int kk = 0; kk < 2; ++kk) {
    const int k0 = kk * 32 + 8 * half;
    v16h ar = load_frag(&xr[btile * 16 + r][k0]);
    v16h ai = load_frag(&xi[btile * 16 + r][k0]);
#pragma unroll
    for (int t = 0; t < 2; ++t) {
      int orow = (og * 2 + t) * 16 + r;
      v16h br  = load_frag(&wr[orow][k0]);
      v16h bi  = load_frag(&wi[orow][k0]);
      v16h bni = load_frag(&wni[orow][k0]);
      accr[t] = wmma_f16(ar, br,  accr[t]);
      accr[t] = wmma_f16(ai, bni, accr[t]);
      acci[t] = wmma_f16(ar, bi,  acci[t]);
      acci[t] = wmma_f16(ai, br,  acci[t]);
    }
  }

#pragma unroll
  for (int t = 0; t < 2; ++t)
#pragma unroll
    for (int v = 0; v < 8; ++v) {
      int b = btile * 16 + 8 * half + v;
      int orow = (og * 2 + t) * 16 + r;
      size_t base = ((size_t)(b * N_ + n) * 64 + orow) * 64;
      XY[base + m]      = (_Float16)accr[t][v];
      XY[base + m + 32] = (_Float16)acci[t][v];
    }
}

// ---------------------------------------------------------------------------
// Kernel 3: truncated inverse DFT.  Per block = one (b,n):
//   out[b,d,n,l] = sum_{k<64} Ycat[d][k] * G[k][l]   ([64x64]x[64x192])
// Y tile (8KB) and Gt table (24KB) are async-staged into LDS; all fragment
// reads are then ds_load_b128.
// ---------------------------------------------------------------------------
__global__ __launch_bounds__(128) void fb_idft(const _Float16* __restrict__ ws,
                                               float* __restrict__ out) {
  __shared__ __align__(16) _Float16 ys[64][72];
  __shared__ __align__(16) _Float16 gs[192][72];
  const int bid = blockIdx.x;
  const int n = bid & (N_ - 1);
  const int b = bid >> 7;
  const int tid = threadIdx.x;
  const int lane = tid & 31;
  const int wave = tid >> 5;     // d-tile
  const int half = lane >> 4;
  const int r = lane & 15;

  const _Float16* Yp = ws + XY_OFF + (size_t)(b * N_ + n) * 4096;  // [d][64]
  const _Float16* Gt = ws + GT_OFF;                                // [l][64]

  // Async-stage Y tile: 64 rows x 8 chunks.
  for (int c = tid; c < 512; c += 128) {
    int row = c >> 3, col = c & 7;
    copy16(Yp + (size_t)row * 64 + col * 8, &ys[row][col * 8]);
  }
  // Async-stage Gt: 192 rows x 8 chunks.
  for (int c = tid; c < 1536; c += 128) {
    int row = c >> 3, col = c & 7;
    copy16(Gt + (size_t)row * 64 + col * 8, &gs[row][col * 8]);
  }
  stage_join();

  v16h a0 = load_frag(&ys[wave * 16 + r][8 * half]);
  v16h a1 = load_frag(&ys[wave * 16 + r][32 + 8 * half]);

  for (int lt = 0; lt < 12; ++lt) {
    v8f acc = {};
    v16h b0 = load_frag(&gs[lt * 16 + r][8 * half]);
    v16h b1 = load_frag(&gs[lt * 16 + r][32 + 8 * half]);
    acc = wmma_f16(a0, b0, acc);
    acc = wmma_f16(a1, b1, acc);
#pragma unroll
    for (int v = 0; v < 8; ++v) {
      int d = wave * 16 + 8 * half + v;
      out[(((size_t)b * D_ + d) * N_ + n) * L_ + lt * 16 + r] = acc[v];
    }
  }
}

extern "C" void kernel_launch(void* const* d_in, const int* in_sizes, int n_in,
                              void* d_out, int out_size, void* d_ws, size_t ws_size,
                              hipStream_t stream) {
  const float* q      = (const float*)d_in[0];
  const float* w_real = (const float*)d_in[1];
  const float* w_imag = (const float*)d_in[2];
  float* out = (float*)d_out;
  _Float16* ws = (_Float16*)d_ws;

  hipLaunchKernelGGL(fb_init_tables, dim3(1), dim3(256), 0, stream, ws);
  hipLaunchKernelGGL(fb_dft,     dim3(B_ * N_), dim3(128), 0, stream, q, ws);
  hipLaunchKernelGGL(fb_modemix, dim3(M_, N_),  dim3(128), 0, stream, w_real, w_imag, ws);
  hipLaunchKernelGGL(fb_idft,    dim3(B_ * N_), dim3(128), 0, stream, ws, out);
}
